// ModelD_26577257627739
// MI455X (gfx1250) — matrix operand order repaired and checked by reading, not compiled, not run on hardware
//
#include <hip/hip_runtime.h>
#include <hip/hip_bf16.h>

// ---------------------------------------------------------------------------
// ModelD: numerator CNN + 9 iterator steps.
//  - trunk (conv1 -> conv2 -> maxpool) fused into ONE kernel, intermediates in
//    LDS, conv2 as bf16 WMMA implicit GEMM over a zero-padded 32x32 LDS image
//    (im2col offsets become compile-time ds immediates).
//  - FC 1296->169/10 as bf16 WMMA GEMM (M=16 samples per wave tile).
//  - depthwise 13x13 + weighted accumulation fused, LDS staged.
//  - gfx1250 async global->LDS staging when available (ASYNCcnt path).
// ---------------------------------------------------------------------------

typedef __attribute__((ext_vector_type(16))) __bf16 v16bf;
typedef __attribute__((ext_vector_type(8)))  __bf16 v8bf;
typedef __attribute__((ext_vector_type(8)))  float  v8f;
typedef __attribute__((ext_vector_type(4)))  int    v4i;

// address-space-qualified int4 pointers for the async global->LDS builtin
typedef v4i __attribute__((address_space(1)))* gv4i_p;   // global int4*
typedef v4i __attribute__((address_space(3)))* lv4i_p;   // LDS int4*

#define B_N      1024
#define IMG      784           // 28*28
#define C1       8
#define C2       16
#define KFC      1296          // 16*9*9
#define KFC_PAD  1312          // 41 * 32
#define KT_FC    41
#define NT_C     11            // ceil(169/16)
#define NPAD_C   176
#define KT_CONV2 7             // ceil(200/32)

#if defined(__HIP_DEVICE_COMPILE__) &&                                        \
    __has_builtin(__builtin_amdgcn_global_load_async_to_lds_b128) &&          \
    __has_builtin(__builtin_amdgcn_s_wait_asynccnt)
#define HAVE_ASYNC_LDS 1
#else
#define HAVE_ASYNC_LDS 0
#endif

#if HAVE_ASYNC_LDS
// per-lane 16B async copy global -> LDS (tracked by ASYNCcnt)
static __device__ __forceinline__ void async_cp16(const void* g, void* l) {
  __builtin_amdgcn_global_load_async_to_lds_b128(
      (gv4i_p)(uintptr_t)g,
      (lv4i_p)(unsigned int)(uintptr_t)l, 0, 0);
}
#endif

static __device__ __forceinline__ v8f wmma_bf16(v16bf a, v16bf b, v8f c) {
  return __builtin_amdgcn_wmma_f32_16x16x32_bf16(false, a, false, b, (short)0, c,
                                                 false, false);
}

// A-fragment K map for 16x32 bf16 A (ISA table): K = (e<8 ? e : e+8) + 8*half
static __device__ __forceinline__ int kmapA(int e, int half) {
  return ((e < 8) ? e : e + 8) + 8 * half;
}

// im2col offset into the zero-padded 32x32 LDS image for reduction index
// kg = ci*25 + u*5 + v  ->  ci*1024 + u*32 + v   (compile-time folded)
static __device__ __forceinline__ constexpr int coff(int kg) {
  return (kg / 25) * 1024 + ((kg % 25) / 5) * 32 + (kg % 5);
}

// ---------------------------------------------------------------------------
// Weight packing (tiny, deterministic, each call).
// ---------------------------------------------------------------------------

// conv2 weights (16,8,5,5) -> WMMA A layout [7 ksteps][32 lanes][16 elems]
__global__ void __launch_bounds__(256)
pack_w2_kernel(const float* __restrict__ w2, __bf16* __restrict__ w2p) {
  int idx = blockIdx.x * blockDim.x + threadIdx.x;   // 7*32*16 = 3584
  if (idx >= KT_CONV2 * 512) return;
  int e = idx & 15, lane = (idx >> 4) & 31, kk = idx >> 9;
  int half = lane >> 4, m = lane & 15;
  int kg = kk * 32 + kmapA(e, half);
  float v = (kg < 200) ? w2[m * 200 + kg] : 0.f;
  w2p[idx] = (__bf16)v;
}

// FC weights (1296,N) -> WMMA B layout [41][NT][32][16]; B K = kmapA(e, half)
__global__ void __launch_bounds__(256)
pack_wl_kernel(const float* __restrict__ wl, __bf16* __restrict__ wlp,
               int N, int NT) {
  int idx = blockIdx.x * blockDim.x + threadIdx.x;   // 41*NT*512
  if (idx >= KT_FC * NT * 512) return;
  int e = idx & 15, lane = (idx >> 4) & 31, g = idx >> 9;  // g = kk*NT + nt
  int kk = g / NT, nt = g % NT;
  int half = lane >> 4, n = nt * 16 + (lane & 15);
  int kg = kk * 32 + kmapA(e, half);
  float v = (kg < KFC && n < N) ? wl[kg * N + n] : 0.f;
  wlp[idx] = (__bf16)v;
}

// ---------------------------------------------------------------------------
// Fused trunk: conv1(1->8,5x5,relu) -> conv2(8->16,5x5,relu; WMMA) -> pool3x3
// One block per sample; all intermediates live in LDS.
// ---------------------------------------------------------------------------
__global__ void __launch_bounds__(256)
trunk_kernel(const float* __restrict__ xin, const float* __restrict__ w1,
             const float* __restrict__ b1, const __bf16* __restrict__ w2p,
             const float* __restrict__ b2, __bf16* __restrict__ feat) {
  __shared__ __align__(16) float  shx[IMG];          // input image (fp32)
  __shared__ float shw1[C1 * 25];
  __shared__ float shb1[C1];
  __shared__ float shb2[C2];
  __shared__ __align__(16) __bf16 sxp[C1 * 1024];    // conv1 out, 32x32 padded
  __shared__ __align__(16) __bf16 sh2[C2 * IMG];     // conv2 out
  const int b = blockIdx.x, tid = threadIdx.x;

  // ---- stage input image ----
#if HAVE_ASYNC_LDS
  if (tid < IMG / 4) async_cp16(xin + (size_t)b * IMG + tid * 4, &shx[tid * 4]);
#else
  for (int i = tid; i < IMG / 4; i += 256)
    ((float4*)shx)[i] = ((const float4*)(xin + (size_t)b * IMG))[i];
#endif
  if (tid < C1 * 25) shw1[tid] = w1[tid];
  if (tid < C1) shb1[tid] = b1[tid];
  if (tid >= 32 && tid < 32 + C2) shb2[tid - 32] = b2[tid - 32];
#if HAVE_ASYNC_LDS
  __builtin_amdgcn_s_wait_asynccnt(0);
#endif
  __syncthreads();

  // ---- conv1 into zero-padded 32x32-per-channel bf16 image ----
  for (int i = tid; i < C1 * 1024; i += 256) {
    const int ci = i >> 10, p = i & 1023;
    const int oy = (p >> 5) - 2, ox = (p & 31) - 2;
    float acc = 0.f;
    if (oy >= 0 && oy < 28 && ox >= 0 && ox < 28) {
      acc = shb1[ci];
#pragma unroll
      for (int u = 0; u < 5; ++u) {
        const int iy = oy + u - 2;
        if (iy < 0 || iy >= 28) continue;
#pragma unroll
        for (int v = 0; v < 5; ++v) {
          const int ix = ox + v - 2;
          if (ix >= 0 && ix < 28)
            acc += shw1[ci * 25 + u * 5 + v] * shx[iy * 28 + ix];
        }
      }
      acc = acc > 0.f ? acc : 0.f;
    }
    sxp[i] = (__bf16)acc;
  }
  __syncthreads();

  // ---- conv2 as implicit GEMM: D[16x784] = W2[16x200] * im2col[200x784] ----
  const int wave = tid >> 5, lane = tid & 31;
  const int half = lane >> 4, mn = lane & 15;
  for (int t = wave; t < 49; t += 8) {               // 49 pixel tiles of 16
    const int n = t * 16 + mn;                       // pixel 0..783
    const int oy = n / 28, ox = n % 28;
    const int boff = oy * 32 + ox;                   // window base in sxp
    v8f acc = {};
#pragma unroll
    for (int kk = 0; kk < KT_CONV2; ++kk) {
      v16bf a = *(const v16bf*)(w2p + (kk * 32 + lane) * 16);
      v16bf bm;
      if (half == 0) {                               // reconverges before wmma
#pragma unroll
        for (int e = 0; e < 16; ++e) {
          const int kg = kk * 32 + kmapA(e, 0);      // compile-time in unroll
          bm[e] = (kg < 200) ? sxp[boff + coff(kg)] : (__bf16)0.f;
        }
      } else {
#pragma unroll
        for (int e = 0; e < 16; ++e) {
          const int kg = kk * 32 + kmapA(e, 1);
          bm[e] = (kg < 200) ? sxp[boff + coff(kg)] : (__bf16)0.f;
        }
      }
      acc = wmma_bf16(a, bm, acc);
    }
#pragma unroll
    for (int v = 0; v < 8; ++v) {                    // D: M = v + 8*half
      const int m = v + 8 * half;
      float o = acc[v] + shb2[m];
      o = o > 0.f ? o : 0.f;
      sh2[m * IMG + n] = (__bf16)o;
    }
  }
  __syncthreads();

  // ---- maxpool 3x3/3 -> feat[b][1312] (zero-padded K for FC) ----
  for (int i = tid; i < KFC_PAD; i += 256) {
    __bf16 r = (__bf16)0.f;
    if (i < KFC) {
      const int c = i / 81, rem = i % 81, py = rem / 9, px = rem % 9;
      const __bf16* base = &sh2[c * IMG + py * 3 * 28 + px * 3];
      float mx = -1e30f;
#pragma unroll
      for (int dy = 0; dy < 3; ++dy)
#pragma unroll
        for (int dx = 0; dx < 3; ++dx) {
          float v = (float)base[dy * 28 + dx];
          mx = v > mx ? v : mx;
        }
      r = (__bf16)mx;
    }
    feat[(size_t)b * KFC_PAD + i] = r;
  }
}

// ---------------------------------------------------------------------------
// FC GEMM: out[B x N] = tanh(feat[B x 1296] * WL[1296 x N] + bias)
// ---------------------------------------------------------------------------
__global__ void __launch_bounds__(256)
fc_wmma_kernel(const __bf16* __restrict__ feat, const __bf16* __restrict__ wlp,
               const float* __restrict__ bias, float* __restrict__ outm,
               int N, int NT) {
  const int tid = threadIdx.x;
  const int widx = blockIdx.x * (blockDim.x >> 5) + (tid >> 5);
  const int mt = widx / NT, nt = widx % NT;
  if (mt >= B_N / 16) return;                        // wave-uniform guard
  const int lane = tid & 31, half = lane >> 4, mn = lane & 15;
  const __bf16* frow = feat + (size_t)(mt * 16 + mn) * KFC_PAD;
  v8f acc = {};
  for (int kk = 0; kk < KT_FC; ++kk) {
    // A: elems 0..7 -> K = kk*32 + 8*half + {0..7}; elems 8..15 -> +16
    v8bf lo = *(const v8bf*)(frow + kk * 32 + 8 * half);
    v8bf hi = *(const v8bf*)(frow + kk * 32 + 16 + 8 * half);
    v16bf a = __builtin_shufflevector(lo, hi, 0, 1, 2, 3, 4, 5, 6, 7,
                                      8, 9, 10, 11, 12, 13, 14, 15);
    v16bf bm = *(const v16bf*)(wlp + ((kk * NT + nt) * 32 + lane) * 16);
    if (kk + 1 < KT_FC)
      __builtin_prefetch(frow + (kk + 1) * 32, 0, 3);  // global_prefetch_b8
    acc = wmma_bf16(a, bm, acc);
  }
#pragma unroll
  for (int v = 0; v < 8; ++v) {
    const int r = mt * 16 + v + 8 * half;
    const int c = nt * 16 + mn;
    if (c < N) {
      float o = tanhf(acc[v] + bias[c]);
      outm[(size_t)r * (NT * 16) + c] = o;
    }
  }
}

// ---------------------------------------------------------------------------
// out = num[:,0] * x ; r0 = x
// ---------------------------------------------------------------------------
__global__ void __launch_bounds__(256)
init_out_kernel(const float* __restrict__ x, const float* __restrict__ num,
                float* __restrict__ out, float* __restrict__ r) {
  int i = blockIdx.x * blockDim.x + threadIdx.x;
  if (i >= B_N * IMG) return;
  float xv = x[i];
  out[i] = num[(i / IMG) * 16] * xv;
  r[i] = xv;
}

// ---------------------------------------------------------------------------
// Depthwise 13x13 correlation over (-1)-padded 40x40 + fused accumulation:
//   rout = corr(pad(rin,-1), w) ; out += num[:,k] * rout
// ---------------------------------------------------------------------------
__global__ void __launch_bounds__(256)
iterstep_kernel(const float* __restrict__ rin, const float* __restrict__ wmat,
                const float* __restrict__ num, int k,
                float* __restrict__ rout, float* __restrict__ out) {
  __shared__ __align__(16) float shr[IMG];           // raw iterate row
  __shared__ __align__(16) float shw[NPAD_C];        // 13x13 kernel (+pad)
  __shared__ float shx[40 * 40];                     // (-1)-padded image
  const int b = blockIdx.x, tid = threadIdx.x;

#if HAVE_ASYNC_LDS
  if (tid < IMG / 4) {
    async_cp16(rin + (size_t)b * IMG + tid * 4, &shr[tid * 4]);
  } else if (tid < IMG / 4 + NPAD_C / 4) {
    const int j = tid - IMG / 4;
    async_cp16(wmat + (size_t)b * NPAD_C + j * 4, &shw[j * 4]);
  }
  __builtin_amdgcn_s_wait_asynccnt(0);
#else
  for (int i = tid; i < IMG / 4; i += 256)
    ((float4*)shr)[i] = ((const float4*)(rin + (size_t)b * IMG))[i];
  if (tid < NPAD_C) shw[tid] = wmat[(size_t)b * NPAD_C + tid];
#endif
  __syncthreads();
  for (int i = tid; i < 1600; i += 256) {
    int iy = i / 40 - 6, ix = i % 40 - 6;
    shx[i] = (iy >= 0 && iy < 28 && ix >= 0 && ix < 28) ? shr[iy * 28 + ix]
                                                        : -1.0f;
  }
  __syncthreads();

  const float nv = num[b * 16 + k];
  for (int p = tid; p < IMG; p += 256) {
    const int oy = p / 28, ox = p % 28;
    float acc = 0.f;
    for (int u = 0; u < 13; ++u) {
      const float* xr = &shx[(oy + u) * 40 + ox];
      const float* wr = &shw[u * 13];
#pragma unroll
      for (int v = 0; v < 13; ++v) acc += wr[v] * xr[v];
    }
    rout[(size_t)b * IMG + p] = acc;
    out[(size_t)b * IMG + p] += nv * acc;
  }
}

// ---------------------------------------------------------------------------
// Host orchestration
// ---------------------------------------------------------------------------
extern "C" void kernel_launch(void* const* d_in, const int* in_sizes, int n_in,
                              void* d_out, int out_size, void* d_ws, size_t ws_size,
                              hipStream_t stream) {
  const float* x    = (const float*)d_in[0];
  const float* n_w1 = (const float*)d_in[1];
  const float* n_b1 = (const float*)d_in[2];
  const float* n_w2 = (const float*)d_in[3];
  const float* n_b2 = (const float*)d_in[4];
  const float* n_wl = (const float*)d_in[5];
  const float* n_bl = (const float*)d_in[6];
  const float* c_w1 = (const float*)d_in[7];
  const float* c_b1 = (const float*)d_in[8];
  const float* c_w2 = (const float*)d_in[9];
  const float* c_b2 = (const float*)d_in[10];
  const float* c_wl = (const float*)d_in[11];
  const float* c_bl = (const float*)d_in[12];
  float* out = (float*)d_out;

  char* wsp = (char*)d_ws;
  auto carve = [&](size_t bytes) -> char* {
    char* p = wsp;
    wsp += (bytes + 255) & ~(size_t)255;
    return p;
  };
  __bf16* feat = (__bf16*)carve((size_t)B_N * KFC_PAD * 2);
  float*  num  = (float*)carve((size_t)B_N * 16 * 4);
  float*  wmat = (float*)carve((size_t)B_N * NPAD_C * 4);
  float*  rA   = (float*)carve((size_t)B_N * IMG * 4);
  float*  rB   = (float*)carve((size_t)B_N * IMG * 4);
  __bf16* w2pN = (__bf16*)carve((size_t)KT_CONV2 * 512 * 2);
  __bf16* w2pC = (__bf16*)carve((size_t)KT_CONV2 * 512 * 2);
  __bf16* wlpN = (__bf16*)carve((size_t)KT_FC * 1 * 512 * 2);
  __bf16* wlpC = (__bf16*)carve((size_t)KT_FC * NT_C * 512 * 2);

  // --- weight packing ---
  pack_w2_kernel<<<(KT_CONV2 * 512 + 255) / 256, 256, 0, stream>>>(n_w2, w2pN);
  pack_w2_kernel<<<(KT_CONV2 * 512 + 255) / 256, 256, 0, stream>>>(c_w2, w2pC);
  pack_wl_kernel<<<(KT_FC * 1 * 512 + 255) / 256, 256, 0, stream>>>(n_wl, wlpN, 10, 1);
  pack_wl_kernel<<<(KT_FC * NT_C * 512 + 255) / 256, 256, 0, stream>>>(c_wl, wlpC, 169, NT_C);

  // --- numerator: num = tanh(trunk(x; n) @ n_wl + n_bl) ---
  trunk_kernel<<<B_N, 256, 0, stream>>>(x, n_w1, n_b1, w2pN, n_b2, feat);
  fc_wmma_kernel<<<(B_N / 16) / 8, 256, 0, stream>>>(feat, wlpN, n_bl, num, 10, 1);

  // --- out = num[:,0]*x ; r = x ---
  init_out_kernel<<<(B_N * IMG + 255) / 256, 256, 0, stream>>>(x, num, out, rA);

  // --- 9 iterator steps: 3 kernels each ---
  float* rc = rA;
  float* rn = rB;
  for (int k = 1; k <= 9; ++k) {
    trunk_kernel<<<B_N, 256, 0, stream>>>(rc, c_w1, c_b1, w2pC, c_b2, feat);
    fc_wmma_kernel<<<((B_N / 16) * NT_C) / 8, 256, 0, stream>>>(feat, wlpC, c_bl,
                                                                wmat, 169, NT_C);
    iterstep_kernel<<<B_N, 256, 0, stream>>>(rc, wmat, num, k, rn, out);
    float* t = rc; rc = rn; rn = t;
  }
}
